// SAGELayer_11587821765008
// MI455X (gfx1250) — matrix operand order repaired and checked by reading, not compiled, not run on hardware
//
#include <hip/hip_runtime.h>

typedef __attribute__((ext_vector_type(16))) _Float16 v16h;
typedef __attribute__((ext_vector_type(8)))  float    v8f;

#define D 64

// ---------------- Kernel 1: zero workspace (agg[N*64] + deg[N]) ----------------
__global__ void zero_ws_kernel(float* __restrict__ p, long long n) {
    long long i = (long long)blockIdx.x * blockDim.x + threadIdx.x;
    const long long stride = (long long)gridDim.x * blockDim.x;
    for (; i < n; i += stride) p[i] = 0.0f;
}

// ---------------- Kernel 2: edge scatter-add (mean aggregator numerator + degree) ----
// One wave (32 lanes) per edge; each lane gathers float2 of x[src] and does two
// non-returning global f32 atomics into agg[dst]. Lane 0 bumps deg[dst].
__global__ void edge_agg_kernel(const float* __restrict__ x,
                                const int*   __restrict__ src,
                                const int*   __restrict__ dst,
                                float*       __restrict__ agg,
                                float*       __restrict__ deg,
                                int E) {
    const long long t = (long long)blockIdx.x * blockDim.x + threadIdx.x;
    const int  lane = (int)(t & 31);
    const long long e = t >> 5;
    if (e >= E) return;
    const int s = src[e];
    const int d = dst[e];
    const int c = lane * 2;
    const float2 v = *(const float2*)(x + (long long)s * D + c);
    float* out = agg + (long long)d * D + c;
    atomicAdd(out + 0, v.x);
    atomicAdd(out + 1, v.y);
    if (lane == 0) atomicAdd(deg + d, 1.0f);
}

// ---------------- Kernel 3: fused mean + concat-GEMM + ReLU + residual GEMM -------
// B-fragment (16-bit, 32x16 KxN): lane holds column C+(lane&15);
// elems 0..7 -> K = kb+0..7, elems 8..15 -> K = kb+16..23 (kb = ko+kb8 from caller).
__device__ inline v16h b_frag(const float* __restrict__ W, int c, int kb) {
    v16h b;
#pragma unroll
    for (int j = 0; j < 16; ++j) b[j] = (_Float16)W[(kb + (j & 7) + ((j >> 3) << 4)) * D + c];
    return b;
}

__global__ void sage_gemm_kernel(const float* __restrict__ x,
                                 const float* __restrict__ agg,
                                 const float* __restrict__ deg,
                                 const float* __restrict__ W_fc,   // [128][64] row-major
                                 const float* __restrict__ b_fc,   // [64]
                                 const float* __restrict__ W_res,  // [64][64]
                                 const float* __restrict__ b_res,  // [64]
                                 float*       __restrict__ out,    // [N][64]
                                 int N) {
    const int lane = threadIdx.x & 31;
    const int wid  = threadIdx.x >> 5;   // 0..7 (8 waves / block)
    const int ct   = wid & 3;            // column tile 0..3 (16 cols each)
    const int rsel = wid >> 2;           // which row tile of the block's pair

    const int c    = ct * 16 + (lane & 15);  // this lane's output column
    const int hi   = lane >> 4;              // lane half
    const int kb8  = hi * 8;                 // A-frag K sub-base
    const int kb16 = hi * 16;                // unused (B-frags also use kb8)

    // Weight B-fragments held in VGPRs for the whole kernel (weights are tiny, L2-hot).
    v16h bX[2], bM[2], bR[2];
#pragma unroll
    for (int ks = 0; ks < 2; ++ks) {
        const int ko = ks * 32;
        bX[ks] = b_frag(W_fc,          c, ko + kb8);  // top half: x part
        bM[ks] = b_frag(W_fc + 64 * D, c, ko + kb8);  // bottom half: mean part
        bR[ks] = b_frag(W_res,         c, ko + kb8);
    }
    const float bf = b_fc[c];
    const float br = b_res[c];

    const int nRowTiles = (N + 15) >> 4;
    for (int rt = blockIdx.x * 2 + rsel; rt < nRowTiles; rt += gridDim.x * 2) {
        const int rowA = rt * 16 + (lane & 15);
        const int rA   = rowA < N ? rowA : N - 1;  // clamped load row
        const float* xrow = x   + (long long)rA * D;
        const float* arow = agg + (long long)rA * D;
        const float  inv  = 1.0f / fmaxf(deg[rA], 1.0f);

        v8f acc_fc  = {};
        v8f acc_res = {};
#pragma unroll
        for (int ks = 0; ks < 2; ++ks) {
            const int ko = ks * 32;
            // A-frag (16-bit, 16x32): lane row = rowA;
            // elems 0..7 -> K = ko+kb8+0..7 ; elems 8..15 -> K = ko+kb8+16..23
            v16h aX, aM;
#pragma unroll
            for (int j = 0; j < 8; ++j) {
                aX[j]     = (_Float16)xrow[ko + kb8 + j];
                aX[j + 8] = (_Float16)xrow[ko + kb8 + 16 + j];
                aM[j]     = (_Float16)(arow[ko + kb8 + j]      * inv);
                aM[j + 8] = (_Float16)(arow[ko + kb8 + 16 + j] * inv);
            }
            acc_fc  = __builtin_amdgcn_wmma_f32_16x16x32_f16(false, aX, false, bX[ks],
                                                             (short)0, acc_fc,  false, false);
            acc_fc  = __builtin_amdgcn_wmma_f32_16x16x32_f16(false, aM, false, bM[ks],
                                                             (short)0, acc_fc,  false, false);
            acc_res = __builtin_amdgcn_wmma_f32_16x16x32_f16(false, aX, false, bR[ks],
                                                             (short)0, acc_res, false, false);
        }

        // C/D layout: VGPR r -> row rt*16 + r + hi*8, column c.
#pragma unroll
        for (int r = 0; r < 8; ++r) {
            const int row = rt * 16 + r + hi * 8;
            if (row < N) {
                const float h = fmaxf(acc_fc[r] + bf, 0.0f) + acc_res[r] + br;
                out[(long long)row * D + c] = h;
            }
        }
    }
}

extern "C" void kernel_launch(void* const* d_in, const int* in_sizes, int n_in,
                              void* d_out, int out_size, void* d_ws, size_t ws_size,
                              hipStream_t stream) {
    const float* x     = (const float*)d_in[0];
    const int*   src   = (const int*)d_in[1];
    const int*   dst   = (const int*)d_in[2];
    const float* W_fc  = (const float*)d_in[3];
    const float* b_fc  = (const float*)d_in[4];
    const float* W_res = (const float*)d_in[5];
    const float* b_res = (const float*)d_in[6];
    float* out = (float*)d_out;

    const int N = in_sizes[0] / D;   // 100000
    const int E = in_sizes[1];       // 800000

    float* agg = (float*)d_ws;                 // N*64 floats
    float* deg = agg + (long long)N * D;       // N floats

    // 1) zero agg + deg
    const long long zn = (long long)N * (D + 1);
    zero_ws_kernel<<<2048, 256, 0, stream>>>(agg, zn);

    // 2) edge scatter-add
    const long long edgeThreads = (long long)E * 32;
    const int edgeBlocks = (int)((edgeThreads + 255) / 256);
    edge_agg_kernel<<<edgeBlocks, 256, 0, stream>>>(x, src, dst, agg, deg, E);

    // 3) fused mean + dual GEMM + ReLU + residual (WMMA)
    sage_gemm_kernel<<<1024, 256, 0, stream>>>(x, agg, deg, W_fc, b_fc, W_res, b_res, out, N);
}
